// get_model_37340445671505
// MI455X (gfx1250) — compile-verified, matrix operand order checked by reading, not run on hardware
//
#include <hip/hip_runtime.h>

// ============================================================================
// MeshSegNet-style forward pass for MI455X (gfx1250, wave32, WMMA).
// All GEMM-shaped work (30 conv layers + adjacency matmuls, ~326 GFLOP total)
// runs through one bf16-WMMA GEMM engine (v_wmma_f32_16x16x32_bf16, f32 acc).
// adjT (64MB) is materialized once and reused by 6 GEMMs out of the 192MB L2.
// ============================================================================

typedef __attribute__((ext_vector_type(16))) __bf16 bf16x16;
typedef __attribute__((ext_vector_type(8)))  __bf16 bf16x8;
typedef __attribute__((ext_vector_type(4)))  __bf16 bf16x4;
typedef __attribute__((ext_vector_type(8)))  float  f32x8;

#define NPTS 4000
#define NGRP 2000
#define KNB  32
#define NPIX (NGRP * KNB)
#define NCLS 17
#define CDIV(a,b) (((a)+(b)-1)/(b))

// ---------------------------------------------------------------------------
// GEMM: C[M,N] = act(alpha[m] * (A[M,K] @ B[K,N]) + beta[m])
// fp32 in/out, bf16 WMMA inside. A,B,C row-major. alpha/beta may be null.
// act: 0=identity, 1=relu, 2=leaky-relu(slope)
// Block = 128 threads (4 waves); tile 64x64x32; each wave owns a 32x32 quadrant
// as 2x2 WMMA accumulators. Fast vectorized fill for interior tiles;
// branch-free clamp+mask fill on edges (no EXEC serialization).
// ---------------------------------------------------------------------------
__global__ __launch_bounds__(128)
void gemm_wmma_bf16(const float* __restrict__ A, int lda,
                    const float* __restrict__ B, int ldb,
                    float* __restrict__ C, int ldc,
                    int M, int N, int K,
                    const float* __restrict__ alpha,
                    const float* __restrict__ beta,
                    int act, float slope)
{
    __shared__ __align__(16) __bf16 sA[64][32];   // [m][k]
    __shared__ __align__(16) __bf16 sB[64][32];   // [n][k]  (B transposed in LDS)

    const int t    = threadIdx.x;
    const int wave = t >> 5;
    const int lane = t & 31;
    const int wr   = (wave >> 1) * 32;     // wave row offset in tile
    const int wc   = (wave & 1) * 32;      // wave col offset in tile
    const int m0   = blockIdx.y * 64;
    const int n0   = blockIdx.x * 64;
    const int row  = lane & 15;            // lane -> M (A) / N (B) within 16
    const int kh   = lane >> 4;            // K-chunk selector (0 or 1)

    const bool mn_full = (m0 + 64 <= M) && (n0 + 64 <= N);

    // fast-path per-thread coordinates (float4 granularity)
    const int a_m  = (t >> 3);             // + 16*i ... wait: see loop (idx>>3)
    const int a_k4 = (t & 7);
    const int b_k  = (t >> 4);
    const int b_n4 = (t & 15);
    (void)a_m; (void)a_k4; (void)b_k; (void)b_n4;

    f32x8 acc[2][2] = {};

    for (int k0 = 0; k0 < K; k0 += 32) {
        if (mn_full && (k0 + 32 <= K)) {
            // ---------------- fast path: unguarded float4 loads ----------------
#pragma unroll
            for (int i = 0; i < 4; ++i) {
                int idx = t + 128 * i;                 // float4 id in A tile
                int mm = idx >> 3, k4 = idx & 7;       // 8 float4 per 32-float row
                const float4 v = *(const float4*)(A + (size_t)(m0 + mm) * lda + k0 + k4 * 4);
                bf16x4 b;
                b[0] = (__bf16)v.x; b[1] = (__bf16)v.y;
                b[2] = (__bf16)v.z; b[3] = (__bf16)v.w;
                *(bf16x4*)&sA[mm][k4 * 4] = b;
            }
#pragma unroll
            for (int i = 0; i < 4; ++i) {
                int idx = t + 128 * i;                 // float4 id in B tile [k][n4]
                int kk = idx >> 4, n4 = idx & 15;      // 16 float4 per 64-float row
                const float4 v = *(const float4*)(B + (size_t)(k0 + kk) * ldb + n0 + n4 * 4);
                sB[n4 * 4 + 0][kk] = (__bf16)v.x;
                sB[n4 * 4 + 1][kk] = (__bf16)v.y;
                sB[n4 * 4 + 2][kk] = (__bf16)v.z;
                sB[n4 * 4 + 3][kk] = (__bf16)v.w;
            }
            if (k0 + 64 <= K) {        // prefetch next k-tile (global_prefetch_b8)
                __builtin_prefetch(A + (size_t)(m0 + (t >> 3)) * lda + k0 + 32 + (t & 7) * 4, 0, 0);
                __builtin_prefetch(B + (size_t)(k0 + 32 + (t >> 4)) * ldb + n0 + (t & 15) * 4, 0, 0);
            }
        } else {
            // -------- edge path: branch-free clamped loads, masked to zero -----
#pragma unroll
            for (int i = 0; i < 16; ++i) {
                int lin = t + 128 * i;
                int kk = lin & 31, mm = lin >> 5;
                int gm = m0 + mm, gk = k0 + kk;
                float msk = (gm < M && gk < K) ? 1.0f : 0.0f;
                int cm = gm < M ? gm : M - 1;
                int ck = gk < K ? gk : K - 1;
                float v = A[(size_t)cm * lda + ck] * msk;
                sA[mm][kk] = (__bf16)v;
            }
#pragma unroll
            for (int i = 0; i < 16; ++i) {
                int lin = t + 128 * i;
                int nn = lin & 63, kk = lin >> 6;
                int gn = n0 + nn, gk = k0 + kk;
                float msk = (gn < N && gk < K) ? 1.0f : 0.0f;
                int cn = gn < N ? gn : N - 1;
                int ck = gk < K ? gk : K - 1;
                float v = B[(size_t)ck * ldb + cn] * msk;
                sB[nn][kk] = (__bf16)v;
            }
        }
        __syncthreads();

        // ---- fragments per ISA 16-bit A layout: lane<16 -> K 0..7 & 16..23,
        //      lane>=16 -> K 8..15 & 24..31 (two contiguous 16B LDS loads) ----
        bf16x16 af[2], bfr[2];
#pragma unroll
        for (int r = 0; r < 2; ++r) {
            const __bf16* p = &sA[wr + r * 16 + row][kh * 8];
            bf16x8 lo = *(const bf16x8*)p;
            bf16x8 hi = *(const bf16x8*)(p + 16);
            af[r] = __builtin_shufflevector(lo, hi, 0,1,2,3,4,5,6,7,8,9,10,11,12,13,14,15);
        }
#pragma unroll
        for (int c = 0; c < 2; ++c) {
            const __bf16* p = &sB[wc + c * 16 + row][kh * 8];
            bf16x8 lo = *(const bf16x8*)p;
            bf16x8 hi = *(const bf16x8*)(p + 16);
            bfr[c] = __builtin_shufflevector(lo, hi, 0,1,2,3,4,5,6,7,8,9,10,11,12,13,14,15);
        }
#pragma unroll
        for (int r = 0; r < 2; ++r)
#pragma unroll
            for (int c = 0; c < 2; ++c)
                acc[r][c] = __builtin_amdgcn_wmma_f32_16x16x32_bf16(
                    false, af[r], false, bfr[c], (short)0, acc[r][c], false, false);
        __syncthreads();
    }

    // ---- epilogue: C/D layout — VGPR v: lanes0-15 row=v, lanes16-31 row=v+8 ----
#pragma unroll
    for (int r = 0; r < 2; ++r)
#pragma unroll
        for (int c = 0; c < 2; ++c)
#pragma unroll
            for (int v = 0; v < 8; ++v) {
                int gm = m0 + wr + r * 16 + v + kh * 8;
                int gn = n0 + wc + c * 16 + row;
                if (gm < M && gn < N) {
                    float x = acc[r][c][v];
                    float a = alpha ? alpha[gm] : 1.0f;
                    float b = beta  ? beta[gm]  : 0.0f;
                    x = a * x + b;
                    if (act == 1)      x = x > 0.0f ? x : 0.0f;
                    else if (act == 2) x = x >= 0.0f ? x : slope * x;
                    C[(size_t)gm * ldc + gn] = x;
                }
            }
}

// ---------------------------------------------------------------------------
// Fold conv bias + batchnorm into per-channel alpha/beta.
// ---------------------------------------------------------------------------
__global__ void fold_bn_k(const float* __restrict__ g, const float* __restrict__ b,
                          const float* __restrict__ mu, const float* __restrict__ var,
                          const float* __restrict__ cb,
                          float* __restrict__ alpha, float* __restrict__ beta,
                          int co, int has_bn)
{
    int i = blockIdx.x * blockDim.x + threadIdx.x;
    if (i >= co) return;
    if (has_bn) {
        float s = g[i] * rsqrtf(var[i] + 1e-5f);
        alpha[i] = s;
        beta[i]  = (cb[i] - mu[i]) * s + b[i];
    } else {
        alpha[i] = 1.0f;
        beta[i]  = cb[i];
    }
}

// ---------------------------------------------------------------------------
// Split point_faces[N][24] into ptsT[12][N] (cols 0..11) and norC[12][N] (12..23)
// ---------------------------------------------------------------------------
__global__ void split_pf_k(const float* __restrict__ pf, float* __restrict__ ptsT,
                           float* __restrict__ norC)
{
    int i = blockIdx.x * blockDim.x + threadIdx.x;
    if (i >= NPTS * 24) return;
    int n = i / 24, c = i % 24;
    float v = pf[i];
    if (c < 12) ptsT[c * NPTS + n] = v;
    else        norC[(c - 12) * NPTS + n] = v;
}

// ---------------------------------------------------------------------------
// Adjacency degree: deg[n] = sum_m any(iface[n]x3 == iface[m]x3)
// ---------------------------------------------------------------------------
__global__ __launch_bounds__(256)
void adj_deg_k(const int* __restrict__ iface, float* __restrict__ deg)
{
    __shared__ int red[256];
    int n = blockIdx.x, t = threadIdx.x;
    int a0 = iface[n * 3], a1 = iface[n * 3 + 1], a2 = iface[n * 3 + 2];
    int cnt = 0;
    for (int m = t; m < NPTS; m += 256) {
        int b0 = iface[m * 3], b1 = iface[m * 3 + 1], b2 = iface[m * 3 + 2];
        bool hit = (a0 == b0) | (a0 == b1) | (a0 == b2) |
                   (a1 == b0) | (a1 == b1) | (a1 == b2) |
                   (a2 == b0) | (a2 == b1) | (a2 == b2);
        cnt += hit ? 1 : 0;
    }
    red[t] = cnt; __syncthreads();
    for (int s = 128; s > 0; s >>= 1) {
        if (t < s) red[t] += red[t + s];
        __syncthreads();
    }
    if (t == 0) deg[n] = (float)red[0];
}

// adjT[m][n] = adj[n][m] = hit(n,m) / deg[n]
__global__ void adj_fill_k(const int* __restrict__ iface, const float* __restrict__ deg,
                           float* __restrict__ adjT)
{
    long long i = (long long)blockIdx.x * blockDim.x + threadIdx.x;
    if (i >= (long long)NPTS * NPTS) return;
    int n = (int)(i % NPTS);         // fast index -> coalesced writes
    int m = (int)(i / NPTS);
    int a0 = iface[n * 3], a1 = iface[n * 3 + 1], a2 = iface[n * 3 + 2];
    int b0 = iface[m * 3], b1 = iface[m * 3 + 1], b2 = iface[m * 3 + 2];
    bool hit = (a0 == b0) | (a0 == b1) | (a0 == b2) |
               (a1 == b0) | (a1 == b1) | (a1 == b2) |
               (a2 == b0) | (a2 == b1) | (a2 == b2);
    adjT[(size_t)m * NPTS + n] = hit ? (1.0f / deg[n]) : 0.0f;
}

// ---------------------------------------------------------------------------
// Farthest point sampling on xyz = point_faces cols 9..11. Single workgroup.
// idx[0] = 0, then 2000 sequential argmax steps (ties -> lowest index).
// ---------------------------------------------------------------------------
__global__ __launch_bounds__(1024)
void fps_k(const float* __restrict__ pf, int* __restrict__ idx_out)
{
    __shared__ float dist[NPTS];
    __shared__ float rv[1024];
    __shared__ int   ri[1024];
    __shared__ float cx, cy, cz;
    __shared__ int   cur;
    int t = threadIdx.x;
    for (int i = t; i < NPTS; i += 1024) dist[i] = 1e10f;
    if (t == 0) cur = 0;
    __syncthreads();
    for (int it = 0; it < NGRP; ++it) {
        if (t == 0) {
            idx_out[it] = cur;
            cx = pf[cur * 24 + 9]; cy = pf[cur * 24 + 10]; cz = pf[cur * 24 + 11];
        }
        __syncthreads();
        float bv = -1.0f; int bi = 0;
        for (int i = t; i < NPTS; i += 1024) {
            float dx = pf[i * 24 + 9]  - cx;
            float dy = pf[i * 24 + 10] - cy;
            float dz = pf[i * 24 + 11] - cz;
            float d  = dx * dx + dy * dy + dz * dz;
            float nd = fminf(dist[i], d);
            dist[i] = nd;
            if (nd > bv) { bv = nd; bi = i; }
        }
        rv[t] = bv; ri[t] = bi; __syncthreads();
        for (int s = 512; s > 0; s >>= 1) {
            if (t < s) {
                if (rv[t + s] > rv[t] || (rv[t + s] == rv[t] && ri[t + s] < ri[t])) {
                    rv[t] = rv[t + s]; ri[t] = ri[t + s];
                }
            }
            __syncthreads();
        }
        if (t == 0) cur = ri[0];
        __syncthreads();
    }
}

__global__ void gather_center_k(const float* __restrict__ pf, const int* __restrict__ fps_idx,
                                float* __restrict__ center, float* __restrict__ cenT)
{
    int i = blockIdx.x * blockDim.x + threadIdx.x;
    if (i >= NGRP * 12) return;
    int g = i / 12, c = i % 12;
    float v = pf[fps_idx[g] * 24 + c];
    center[g * 12 + c] = v;
    cenT[c * NGRP + g] = v;
}

// ---------------------------------------------------------------------------
// 32-NN per group in 12-D (ties -> lowest index): iterative argmin extraction.
// ---------------------------------------------------------------------------
__global__ __launch_bounds__(256)
void knn_k(const float* __restrict__ pf, const float* __restrict__ center,
           int* __restrict__ knn_idx)
{
    __shared__ float d[NPTS];
    __shared__ float rv[256];
    __shared__ int   ri[256];
    int g = blockIdx.x, t = threadIdx.x;
    float cc[12];
#pragma unroll
    for (int j = 0; j < 12; ++j) cc[j] = center[g * 12 + j];
    for (int i = t; i < NPTS; i += 256) {
        float s = 0.0f;
#pragma unroll
        for (int j = 0; j < 12; ++j) {
            float df = pf[i * 24 + j] - cc[j];
            s += df * df;
        }
        d[i] = s;
    }
    __syncthreads();
    for (int sel = 0; sel < KNB; ++sel) {
        float bv = 1e30f; int bi = 0;
        for (int i = t; i < NPTS; i += 256)
            if (d[i] < bv) { bv = d[i]; bi = i; }
        rv[t] = bv; ri[t] = bi; __syncthreads();
        for (int s = 128; s > 0; s >>= 1) {
            if (t < s) {
                if (rv[t + s] < rv[t] || (rv[t + s] == rv[t] && ri[t + s] < ri[t])) {
                    rv[t] = rv[t + s]; ri[t] = ri[t + s];
                }
            }
            __syncthreads();
        }
        if (t == 0) { knn_idx[g * KNB + sel] = ri[0]; d[ri[0]] = 1e30f; }
        __syncthreads();
    }
}

// ---------------------------------------------------------------------------
// Build group feature maps, pixel index p = g*32+k:
//   nbp[c][p]   = pts[knn]           (uncentered neighbor)
//   nbc[c][p]   = pts[knn] - center  (centered, = _group's nb)
//   delta[c][p] = center - nbc  (c<12), nbc (c>=12)   [matches reference]
// ---------------------------------------------------------------------------
__global__ void build_group_k(const float* __restrict__ pf, const int* __restrict__ knn,
                              const float* __restrict__ center,
                              float* __restrict__ nbp, float* __restrict__ nbc,
                              float* __restrict__ delta)
{
    int i = blockIdx.x * blockDim.x + threadIdx.x;
    if (i >= NPIX * 12) return;
    int c = i % 12, p = i / 12;
    int g = p / KNB;
    float pv = pf[knn[p] * 24 + c];
    float cv = center[g * 12 + c];
    float ctr = pv - cv;
    nbp[c * NPIX + p] = pv;
    nbc[c * NPIX + p] = ctr;
    delta[c * NPIX + p]        = cv - ctr;
    delta[(c + 12) * NPIX + p] = ctr;
}

// ---------------------------------------------------------------------------
// Graph attention: softmax over k of e[c][g*32+k], weighted sum of feat.
// ---------------------------------------------------------------------------
__global__ void graph_attn_k(const float* __restrict__ e, const float* __restrict__ feat,
                             float* __restrict__ gf, int C)
{
    int i = blockIdx.x * blockDim.x + threadIdx.x;
    if (i >= C * NGRP) return;
    int c = i / NGRP, g = i % NGRP;
    const float* er = &e[(size_t)c * NPIX + g * KNB];
    const float* fr = &feat[(size_t)c * NPIX + g * KNB];
    float mx = -1e30f;
#pragma unroll
    for (int k = 0; k < KNB; ++k) mx = fmaxf(mx, er[k]);
    float s = 0.0f, acc = 0.0f;
#pragma unroll
    for (int k = 0; k < KNB; ++k) {
        float w = __expf(er[k] - mx);
        s += w; acc += w * fr[k];
    }
    gf[c * NGRP + g] = acc / s;
}

__global__ void maxpool_k_k(const float* __restrict__ in, float* __restrict__ out, int C)
{
    int i = blockIdx.x * blockDim.x + threadIdx.x;
    if (i >= C * NGRP) return;
    int c = i / NGRP, g = i % NGRP;
    const float* r = &in[(size_t)c * NPIX + g * KNB];
    float mx = -1e30f;
#pragma unroll
    for (int k = 0; k < KNB; ++k) mx = fmaxf(mx, r[k]);
    out[c * NGRP + g] = mx;
}

// cat[256][NPIX]: c<128 -> fg[c][g] broadcast, else f[c-128][p]
__global__ void enc_concat_k(const float* __restrict__ fg, const float* __restrict__ f,
                             float* __restrict__ cat)
{
    long long i = (long long)blockIdx.x * blockDim.x + threadIdx.x;
    if (i >= (long long)256 * NPIX) return;
    int c = (int)(i / NPIX), p = (int)(i % NPIX);
    int g = p / KNB;
    cat[i] = (c < 128) ? fg[c * NGRP + g] : f[(size_t)(c - 128) * NPIX + p];
}

// xg[c] = max over g, xg[C+c] = mean over g
__global__ void pos_pool_k(const float* __restrict__ pos, float* __restrict__ xg, int C)
{
    int c = blockIdx.x * blockDim.x + threadIdx.x;
    if (c >= C) return;
    float mx = -1e30f, s = 0.0f;
    for (int g = 0; g < NGRP; ++g) {
        float v = pos[(size_t)c * NGRP + g];
        mx = fmaxf(mx, v); s += v;
    }
    xg[c] = mx;
    xg[C + c] = s / (float)NGRP;
}

// ---------------------------------------------------------------------------
// 3-NN of each point among the 2000 centers (12-D), inverse-distance weights.
// ---------------------------------------------------------------------------
__global__ void top3_k(const float* __restrict__ pf, const float* __restrict__ center,
                       int* __restrict__ idx3, float* __restrict__ w3)
{
    int n = blockIdx.x * blockDim.x + threadIdx.x;
    if (n >= NPTS) return;
    float p[12];
#pragma unroll
    for (int j = 0; j < 12; ++j) p[j] = pf[n * 24 + j];
    float bd0 = 1e30f, bd1 = 1e30f, bd2 = 1e30f;
    int   bi0 = 0, bi1 = 0, bi2 = 0;
    for (int g = 0; g < NGRP; ++g) {
        float s = 0.0f;
#pragma unroll
        for (int j = 0; j < 12; ++j) {
            float df = p[j] - center[g * 12 + j];
            s += df * df;
        }
        if (s < bd0)      { bd2 = bd1; bi2 = bi1; bd1 = bd0; bi1 = bi0; bd0 = s; bi0 = g; }
        else if (s < bd1) { bd2 = bd1; bi2 = bi1; bd1 = s;  bi1 = g; }
        else if (s < bd2) { bd2 = s;  bi2 = g; }
    }
    float w0 = 1.0f / (bd0 + 1e-8f);
    float w1 = 1.0f / (bd1 + 1e-8f);
    float w2 = 1.0f / (bd2 + 1e-8f);
    float s  = w0 + w1 + w2;
    idx3[n * 3 + 0] = bi0; idx3[n * 3 + 1] = bi1; idx3[n * 3 + 2] = bi2;
    w3[n * 3 + 0] = w0 / s; w3[n * 3 + 1] = w1 / s; w3[n * 3 + 2] = w2 / s;
}

// out[c][n] = sum_j feat[c][idx3[n][j]] * w3[n][j]
__global__ void interp3_k(const float* __restrict__ feat, const int* __restrict__ idx3,
                          const float* __restrict__ w3, float* __restrict__ out, int C)
{
    long long i = (long long)blockIdx.x * blockDim.x + threadIdx.x;
    if (i >= (long long)C * NPTS) return;
    int c = (int)(i / NPTS), n = (int)(i % NPTS);
    const float* fr = &feat[(size_t)c * NGRP];
    out[i] = fr[idx3[n * 3 + 0]] * w3[n * 3 + 0]
           + fr[idx3[n * 3 + 1]] * w3[n * 3 + 1]
           + fr[idx3[n * 3 + 2]] * w3[n * 3 + 2];
}

// cat[1024][N]: rows 0..511 = f0, rows 512..1023 = xg[c-512] broadcast
__global__ void concat_f0_xg_k(const float* __restrict__ f0, const float* __restrict__ xg,
                               float* __restrict__ out)
{
    long long i = (long long)blockIdx.x * blockDim.x + threadIdx.x;
    if (i >= (long long)1024 * NPTS) return;
    int c = (int)(i / NPTS);
    out[i] = (c < 512) ? f0[i] : xg[c - 512];
}

// out = x * sigmoid(xl+xg) + y * (1 - sigmoid(xl+xg))
__global__ void aff_mix_k(const float* __restrict__ x, const float* __restrict__ y,
                          const float* __restrict__ xl, const float* __restrict__ xg,
                          float* __restrict__ out)
{
    long long i = (long long)blockIdx.x * blockDim.x + threadIdx.x;
    if (i >= (long long)1024 * NPTS) return;
    float s = 1.0f / (1.0f + __expf(-(xl[i] + xg[i])));
    out[i] = x[i] * s + y[i] * (1.0f - s);
}

// out[n][c] = log_softmax over channel of logits[c][n]
__global__ void logsoftmax_k(const float* __restrict__ logits, float* __restrict__ out)
{
    int n = blockIdx.x * blockDim.x + threadIdx.x;
    if (n >= NPTS) return;
    float mx = -1e30f;
#pragma unroll
    for (int c = 0; c < NCLS; ++c) mx = fmaxf(mx, logits[c * NPTS + n]);
    float s = 0.0f;
#pragma unroll
    for (int c = 0; c < NCLS; ++c) s += __expf(logits[c * NPTS + n] - mx);
    float ls = __logf(s) + mx;
#pragma unroll
    for (int c = 0; c < NCLS; ++c) out[n * NCLS + c] = logits[c * NPTS + n] - ls;
}

// ============================================================================
// Host orchestration
// ============================================================================
enum { Ln1 = 0, Ln2, Ln3, Ln4, Lc1, Lc2, Lc3, Lc4, La1, La2, La3,
       Lencf1, Lencf2, Lencs1, Lencs2, Lpos1, Lpos2,
       Lp0a, Lp0b, Lp1a, Lp1b, Laffl1, Laffl2, Laffg1, Laffg2,
       Lfa, Ls2, Ls3, Ls4, Lpred };

static const int CI[30] = {12,64,128,448, 12,64,128,1024, 24,24,24,
                           12,64,256,256, 12,128,
                           524,512,460,448, 1024,256,1024,256,
                           1024,1024,512,256,128};
static const int CO[30] = {64,128,256,512, 64,128,256,512, 64,128,256,
                           64,128,256,512, 128,256,
                           512,512,448,1024, 256,1024,256,1024,
                           1024,512,256,128,17};
static const int HASBN[30] = {1,1,1,1, 1,1,1,1, 1,1,1,
                              1,0,1,0, 1,1,
                              1,1,1,1, 1,1,1,1,
                              1,1,1,1,0};

extern "C" void kernel_launch(void* const* d_in, const int* in_sizes, int n_in,
                              void* d_out, int out_size, void* d_ws, size_t ws_size,
                              hipStream_t stream)
{
    (void)in_sizes; (void)n_in; (void)out_size; (void)ws_size;
    const float* pf    = (const float*)d_in[0];
    const int*   iface = (const int*)d_in[1];
    auto PP = [&](int li, int j) { return (const float*)d_in[2 + li * 6 + j]; };
    // param order per layer: W=0, cb=1, g=2, b=3, m=4, v=5

    // ---- bump allocator over d_ws ----
    size_t off = 0;
    auto alloc = [&](size_t nfloats) -> float* {
        size_t cur = (off + 255) & ~(size_t)255;
        off = cur + nfloats * sizeof(float);
        return (float*)((char*)d_ws + cur);
    };

    float* alphas  = alloc(30 * 1024);
    float* betas   = alloc(30 * 1024);
    float* adjT    = alloc((size_t)NPTS * NPTS);
    float* adj2T   = alloc((size_t)NPTS * NPTS);
    float* deg     = alloc(NPTS);
    float* ptsT    = alloc(12 * NPTS);
    float* norC    = alloc(12 * NPTS);
    float* tmpA    = alloc(256 * NPTS);
    float* n123    = alloc(448 * NPTS);
    float* norfpre = alloc(448 * NPTS);
    float* h_in    = alloc(1024 * NPTS);      // rows 0..511 coor, 512..1023 nor_f
    int*   fps_idx = (int*)alloc(NGRP);
    float* center  = alloc(NGRP * 12);
    float* cenT    = alloc(12 * NGRP);
    int*   knn     = (int*)alloc(NGRP * KNB);
    float* nbp     = alloc((size_t)12 * NPIX);
    float* nbc     = alloc((size_t)12 * NPIX);
    float* delta   = alloc((size_t)24 * NPIX);
    float* coor1   = alloc((size_t)64 * NPIX);
    float* e_buf   = alloc((size_t)256 * NPIX);
    float* coor2   = alloc((size_t)128 * NPIX);
    float* coor3   = alloc((size_t)256 * NPIX);
    float* corf    = alloc(448 * NGRP);
    float* s2buf   = alloc((size_t)512 * NPIX);
    float* tokens  = alloc(512 * NGRP);
    float* fg      = alloc(128 * NGRP);
    float* posA    = alloc(128 * NGRP);
    float* posB    = alloc(256 * NGRP);
    float* xg      = alloc(512);
    int*   idx3    = (int*)alloc(NPTS * 3);
    float* w3      = alloc(NPTS * 3);
    float* new524  = alloc(524 * NPTS);
    float* f0mid   = alloc(512 * NPTS);
    float* f0      = alloc(512 * NPTS);
    float* new460  = alloc(460 * NPTS);
    float* p1mid   = alloc(448 * NPTS);
    float* p1out   = alloc(1024 * NPTS);
    float* f1buf   = alloc(1024 * NPTS);
    float* catG    = alloc(1024 * NPTS);
    float* xglob   = alloc(1024 * NPTS);
    float* affT    = alloc(256 * NPTS);
    float* xl      = alloc(1024 * NPTS);
    float* xgl     = alloc(1024 * NPTS);
    float* h4      = alloc(128 * NPTS);
    float* logits  = alloc(NCLS * NPTS);
    // reuse of big transient buffers:
    float* encf1 = coor1;   // [64][NPIX]
    float* encf  = coor2;   // [128][NPIX]
    float* cat   = coor3;   // [256][NPIX]
    float* s1    = e_buf;   // [256][NPIX]
    float* affout = catG;   // [1024][N]
    float* amulA  = p1out;  // [1024][N]
    float* hbuf   = xl;     // [1024][N]
    float* h2     = xgl;    // [512][N]
    float* h3     = affT;   // [256][N]

    // ---- GEMM launch helpers ----
    auto conv = [&](int li, const float* X, int n, float* Y, int act, float slope) {
        dim3 grid(CDIV(n, 64), CDIV(CO[li], 64));
        gemm_wmma_bf16<<<grid, 128, 0, stream>>>(PP(li, 0), CI[li], X, n, Y, n,
                                                 CO[li], n, CI[li],
                                                 alphas + li * 1024, betas + li * 1024,
                                                 act, slope);
    };
    auto mm = [&](const float* A, const float* B, float* C, int M, int n, int K) {
        dim3 grid(CDIV(n, 64), CDIV(M, 64));
        gemm_wmma_bf16<<<grid, 128, 0, stream>>>(A, K, B, n, C, n, M, n, K,
                                                 nullptr, nullptr, 0, 0.0f);
    };

    // ---- fold BN+bias into alpha/beta for all 30 layers ----
    for (int li = 0; li < 30; ++li)
        fold_bn_k<<<CDIV(CO[li], 256), 256, 0, stream>>>(
            PP(li, 2), PP(li, 3), PP(li, 4), PP(li, 5), PP(li, 1),
            alphas + li * 1024, betas + li * 1024, CO[li], HASBN[li]);

    // ---- input splits ----
    split_pf_k<<<CDIV(NPTS * 24, 256), 256, 0, stream>>>(pf, ptsT, norC);

    // ---- adjacency: adjT then adj2T = adjT @ adjT ----
    adj_deg_k<<<NPTS, 256, 0, stream>>>(iface, deg);
    adj_fill_k<<<CDIV(NPTS * NPTS, 256), 256, 0, stream>>>(iface, deg, adjT);
    mm(adjT, adjT, adj2T, NPTS, NPTS, NPTS);               // (adj@adj)^T

    // ---- normal branch ----
    conv(Ln1, norC, NPTS, tmpA, 2, 0.2f);
    mm(tmpA, adjT, n123, 64, NPTS, NPTS);                  // n1
    conv(Ln2, n123, NPTS, tmpA, 2, 0.2f);
    mm(tmpA, adjT, n123 + (size_t)64 * NPTS, 128, NPTS, NPTS);   // n2
    conv(Ln3, n123 + (size_t)64 * NPTS, NPTS, tmpA, 2, 0.2f);
    mm(tmpA, adjT, n123 + (size_t)192 * NPTS, 256, NPTS, NPTS);  // n3
    mm(n123, adj2T, norfpre, 448, NPTS, NPTS);             // amul(adj2, concat)
    conv(Ln4, norfpre, NPTS, h_in + (size_t)512 * NPTS, 2, 0.2f); // nor_f

    // ---- grouping ----
    fps_k<<<1, 1024, 0, stream>>>(pf, fps_idx);
    gather_center_k<<<CDIV(NGRP * 12, 256), 256, 0, stream>>>(pf, fps_idx, center, cenT);
    knn_k<<<NGRP, 256, 0, stream>>>(pf, center, knn);
    build_group_k<<<CDIV(NPIX * 12, 256), 256, 0, stream>>>(pf, knn, center, nbp, nbc, delta);

    // ---- coordinate branch with graph attention ----
    conv(Lc1, nbp, NPIX, coor1, 2, 0.2f);
    conv(La1, delta, NPIX, e_buf, 2, 0.2f);
    graph_attn_k<<<CDIV(64 * NGRP, 256), 256, 0, stream>>>(e_buf, coor1, corf, 64);
    conv(Lc2, coor1, NPIX, coor2, 2, 0.2f);
    conv(La2, delta, NPIX, e_buf, 2, 0.2f);
    graph_attn_k<<<CDIV(128 * NGRP, 256), 256, 0, stream>>>(e_buf, coor2, corf + 64 * NGRP, 128);
    conv(Lc3, coor2, NPIX, coor3, 2, 0.2f);
    conv(La3, delta, NPIX, e_buf, 2, 0.2f);
    graph_attn_k<<<CDIV(256 * NGRP, 256), 256, 0, stream>>>(e_buf, coor3, corf + 192 * NGRP, 256);

    // ---- encoder -> tokens ----
    conv(Lencf1, nbc, NPIX, encf1, 2, 0.2f);
    conv(Lencf2, encf1, NPIX, encf, 0, 0.0f);              // conv + bias only
    maxpool_k_k<<<CDIV(128 * NGRP, 256), 256, 0, stream>>>(encf, fg, 128);
    enc_concat_k<<<CDIV(256 * NPIX, 256), 256, 0, stream>>>(fg, encf, cat);
    conv(Lencs1, cat, NPIX, s1, 2, 0.2f);
    conv(Lencs2, s1, NPIX, s2buf, 0, 0.0f);
    maxpool_k_k<<<CDIV(512 * NGRP, 256), 256, 0, stream>>>(s2buf, tokens, 512);

    // ---- positional global feature ----
    conv(Lpos1, cenT, NGRP, posA, 2, 0.2f);
    conv(Lpos2, posA, NGRP, posB, 2, 0.2f);
    pos_pool_k<<<1, 256, 0, stream>>>(posB, xg, 256);

    // ---- feature propagation (shared 3-NN / weights) ----
    top3_k<<<CDIV(NPTS, 256), 256, 0, stream>>>(pf, center, idx3, w3);
    hipMemcpyAsync(new524, ptsT, (size_t)12 * NPTS * sizeof(float),
                   hipMemcpyDeviceToDevice, stream);
    interp3_k<<<CDIV(512 * NPTS, 256), 256, 0, stream>>>(tokens, idx3, w3,
                                                         new524 + (size_t)12 * NPTS, 512);
    conv(Lp0a, new524, NPTS, f0mid, 1, 0.0f);
    conv(Lp0b, f0mid, NPTS, f0, 1, 0.0f);
    hipMemcpyAsync(new460, ptsT, (size_t)12 * NPTS * sizeof(float),
                   hipMemcpyDeviceToDevice, stream);
    interp3_k<<<CDIV(448 * NPTS, 256), 256, 0, stream>>>(corf, idx3, w3,
                                                         new460 + (size_t)12 * NPTS, 448);
    conv(Lp1a, new460, NPTS, p1mid, 1, 0.0f);
    conv(Lp1b, p1mid, NPTS, p1out, 1, 0.0f);
    mm(p1out, adjT, f1buf, 1024, NPTS, NPTS);              // f1 = amul(adj, ...)

    concat_f0_xg_k<<<CDIV(1024 * NPTS, 256), 256, 0, stream>>>(f0, xg, catG);
    mm(catG, adjT, xglob, 1024, NPTS, NPTS);               // x_global

    // ---- affinity fusion ----
    conv(Laffl1, f1buf, NPTS, affT, 2, 0.01f);
    conv(Laffl2, affT, NPTS, xl, 0, 0.0f);                 // conv+bn, no act
    conv(Laffg1, xglob, NPTS, affT, 2, 0.01f);
    conv(Laffg2, affT, NPTS, xgl, 0, 0.0f);
    aff_mix_k<<<CDIV(1024 * NPTS, 256), 256, 0, stream>>>(f1buf, xglob, xl, xgl, affout);
    mm(affout, adj2T, amulA, 1024, NPTS, NPTS);            // amul(adj2, aff)
    conv(Lc4, amulA, NPTS, h_in, 2, 0.2f);                 // coor -> rows 0..511

    // ---- head ----
    conv(Lfa, h_in, NPTS, hbuf, 2, 0.2f);
    conv(Ls2, hbuf, NPTS, h2, 2, 0.2f);
    conv(Ls3, h2, NPTS, h3, 2, 0.2f);
    conv(Ls4, h3, NPTS, h4, 2, 0.2f);
    conv(Lpred, h4, NPTS, logits, 0, 0.0f);
    logsoftmax_k<<<CDIV(NPTS, 256), 256, 0, stream>>>(logits, (float*)d_out);
}
